// MultiHeadAttention_28939489641225
// MI455X (gfx1250) — compile-verified, matrix operand order checked by reading, not run on hardware
//
#include <hip/hip_runtime.h>
#include <hip/hip_bf16.h>

#define NN   65536
#define DD   256
#define EE   128
#define HH   8
#define HIDW 256
#define GG   128

typedef __attribute__((ext_vector_type(16))) _Float16 v16h;
typedef __attribute__((ext_vector_type(8)))  _Float16 v8h_t;
typedef __attribute__((ext_vector_type(8)))  float    v8f;

#if defined(__gfx1250__) && __has_builtin(__builtin_amdgcn_global_load_async_to_lds_b128)
#define ASYNC_LDS 1
#endif

// ---------------------------------------------------------------------------
// Async global->LDS 16B copy (gfx1250 GLOBAL_LOAD_ASYNC_TO_LDS_B128, ASYNCcnt)
// with synchronous fallback. Builtin signature (verified by compiler):
//   void(int4 AS1*, int4 AS3*, imm int offset, imm int cpol)
// ---------------------------------------------------------------------------
#ifdef ASYNC_LDS
typedef int v4i_raw __attribute__((vector_size(4 * sizeof(int))));
typedef __attribute__((address_space(1))) v4i_raw as1_v4i;
typedef __attribute__((address_space(3))) v4i_raw as3_v4i;
#endif

__device__ __forceinline__ void copy16B_g2l(_Float16* ldst, const _Float16* gsrc) {
#ifdef ASYNC_LDS
  __builtin_amdgcn_global_load_async_to_lds_b128(
      (as1_v4i*)(unsigned long long)gsrc,
      (as3_v4i*)(unsigned int)(unsigned long long)ldst,
      0, 0);
#else
  *(v8h_t*)ldst = *(const v8h_t*)gsrc;
#endif
}

__device__ __forceinline__ void copy_fence() {
#ifdef ASYNC_LDS
#if __has_builtin(__builtin_amdgcn_s_wait_asynccnt)
  __builtin_amdgcn_s_wait_asynccnt(0);
#else
  asm volatile("s_wait_asynccnt 0x0" ::: "memory");
#endif
#endif
}

// ---------------------------------------------------------------------------
// WMMA helpers (CDNA5 gfx1250, wave32).
// A frag (16x32 f16): lane l holds row m=l&15; k-chunks [8*(l>>4) .. +7] and
// [16+8*(l>>4) .. +7] -> two contiguous 16B loads from a row-major tile.
// B frag (32x16): symmetric (lane = n); weights pre-transposed to [out,in].
// C/D frag (16x16 f32): lane l holds column n=l&15, rows m=(l>>4)*8+r.
// ---------------------------------------------------------------------------
__device__ __forceinline__ v16h load_frag(const _Float16* p, int ld) {
  int lane = threadIdx.x & 31;
  int row  = lane & 15;
  int grp  = lane >> 4;
  const _Float16* r = p + (size_t)row * ld + grp * 8;
  v8h_t c0 = *(const v8h_t*)(r);
  v8h_t c1 = *(const v8h_t*)(r + 16);
  v16h out;
  #pragma unroll
  for (int i = 0; i < 8; i++) { out[i] = c0[i]; out[i + 8] = c1[i]; }
  return out;
}

__device__ __forceinline__ v8f wmma_f16(v16h a, v16h b, v8f c) {
  return __builtin_amdgcn_wmma_f32_16x16x32_f16(false, a, false, b,
                                                (short)0, c, false, false);
}

__device__ __forceinline__ void store_tile_f16(_Float16* p, int ld, v8f acc,
                                               bool relu) {
  int lane  = threadIdx.x & 31;
  int col   = lane & 15;
  int rbase = (lane >> 4) * 8;
  #pragma unroll
  for (int r = 0; r < 8; r++) {
    float v = acc[r];
    if (relu) v = v > 0.f ? v : 0.f;
    p[(size_t)(rbase + r) * ld + col] = (_Float16)v;
  }
}

// ---------------------------------------------------------------------------
// K0: sm = score_matrix + pos   (f32 -> f16), vectorized by 8
// ---------------------------------------------------------------------------
__global__ void prep_sm(const float* __restrict__ sm, const float* __restrict__ pos,
                        _Float16* __restrict__ out, int total8) {
  int i = blockIdx.x * blockDim.x + threadIdx.x;
  if (i >= total8) return;
  size_t base = (size_t)i * 8;
  int d = (int)(base & (DD - 1));
  v8h_t o;
  #pragma unroll
  for (int j = 0; j < 8; j++) o[j] = (_Float16)(sm[base + j] + pos[d + j]);
  *(v8h_t*)(out + base) = o;
}

// ---------------------------------------------------------------------------
// K0b: weight convert + transpose  [B,R,C] f32 -> [B,C,R] f16
// ---------------------------------------------------------------------------
__global__ void cvt_transpose(const float* __restrict__ in, _Float16* __restrict__ out,
                              int B, int R, int C) {
  int i = blockIdx.x * blockDim.x + threadIdx.x;
  int total = B * R * C;
  if (i >= total) return;
  int b = i / (R * C);
  int rem = i - b * R * C;
  int r = rem / C;
  int c = rem - r * C;
  out[(size_t)b * R * C + (size_t)c * R + r] = (_Float16)in[i];
}

// ---------------------------------------------------------------------------
// K1: fused per-head 2-layer MLP (q/k/v), all WMMA.
// grid = (N/64, 24)  blockIdx.y = proj*8+head.  256 threads = 8 waves.
// Stage A: 64x256 = relu(sm[64x256] @ W1t^T + b1) into LDS
// Stage B: 64x128 = h1 @ W2t^T + b2  -> q/k/v f16 global [N, H*E]
// ---------------------------------------------------------------------------
__global__ __launch_bounds__(256)
void qkv_kernel(const _Float16* __restrict__ smh,
                const _Float16* w1q, const _Float16* w1k, const _Float16* w1v,
                const float* b1q, const float* b1k, const float* b1v,
                const _Float16* w2q, const _Float16* w2k, const _Float16* w2v,
                const float* b2q, const float* b2k, const float* b2v,
                _Float16* qb, _Float16* kb, _Float16* vb) {
  __shared__ _Float16 smt[64 * 256];  // 32 KB
  __shared__ _Float16 h1 [64 * 256];  // 32 KB
  int tid  = threadIdx.x;
  int n0   = blockIdx.x * 64;
  int y    = blockIdx.y;
  int proj = y >> 3;
  int head = y & 7;
  const _Float16* W1t = (proj == 0 ? w1q : (proj == 1 ? w1k : w1v)) + head * HIDW * DD;
  const float*    b1  = (proj == 0 ? b1q : (proj == 1 ? b1k : b1v)) + head * HIDW;
  const _Float16* W2t = (proj == 0 ? w2q : (proj == 1 ? w2k : w2v)) + head * EE * HIDW;
  const float*    b2  = (proj == 0 ? b2q : (proj == 1 ? b2k : b2v)) + head * EE;
  _Float16* outb = (proj == 0 ? qb : (proj == 1 ? kb : vb));

  // async-stage the 64x256 activation tile (2048 x 16B chunks)
  for (int i = tid; i < 64 * 32; i += 256) {
    int r = i >> 5, c8 = (i & 31) << 3;
    copy16B_g2l(&smt[r * 256 + c8], &smh[((size_t)(n0 + r)) * DD + c8]);
  }
  copy_fence();
  __syncthreads();

  int w = tid >> 5, lane = tid & 31, l15 = lane & 15;
  int mt = w >> 1;  // 4 row-tiles of 16, 2 waves each

  // -------- Stage A: K=256, out cols 256, relu --------
  {
    int cbase = (w & 1) * 128;
    for (int ct = 0; ct < 8; ct++) {
      int col0 = cbase + ct * 16;
      if (ct < 7) __builtin_prefetch(&W1t[(col0 + 16) * DD], 0, 0);
      float bias = b1[col0 + l15];
      v8f acc;
      #pragma unroll
      for (int i = 0; i < 8; i++) acc[i] = bias;
      #pragma unroll
      for (int kk = 0; kk < 8; kk++) {
        v16h a = load_frag(&smt[mt * 16 * 256 + kk * 32], 256);
        v16h b = load_frag(&W1t[col0 * DD + kk * 32], DD);
        acc = wmma_f16(a, b, acc);
      }
      store_tile_f16(&h1[mt * 16 * 256 + col0], 256, acc, true);
    }
  }
  __syncthreads();

  // -------- Stage B: K=256, out cols 128 --------
  {
    int cbase = (w & 1) * 64;
    for (int ct = 0; ct < 4; ct++) {
      int col0 = cbase + ct * 16;
      if (ct < 3) __builtin_prefetch(&W2t[(col0 + 16) * HIDW], 0, 0);
      float bias = b2[col0 + l15];
      v8f acc;
      #pragma unroll
      for (int i = 0; i < 8; i++) acc[i] = bias;
      #pragma unroll
      for (int kk = 0; kk < 8; kk++) {
        v16h a = load_frag(&h1[mt * 16 * 256 + kk * 32], 256);
        v16h b = load_frag(&W2t[col0 * HIDW + kk * 32], HIDW);
        acc = wmma_f16(a, b, acc);
      }
      store_tile_f16(outb + ((size_t)(n0 + mt * 16)) * (HH * EE) + head * EE + col0,
                     HH * EE, acc, false);
    }
  }
}

// ---------------------------------------------------------------------------
// K2: per-node scores[n,h,g] = q[n,h]·k[n,g] / sqrt(E)    (tiny; VALU)
// vectorized: 16-byte LDS loads, 8 MACs each
// ---------------------------------------------------------------------------
__global__ __launch_bounds__(64)
void scores_kernel(const _Float16* __restrict__ qb, const _Float16* __restrict__ kb,
                   float* __restrict__ scores) {
  int n = blockIdx.x;
  __shared__ _Float16 qs[HH * EE], ks[HH * EE];
  int tid = threadIdx.x;
  for (int i = tid; i < 128; i += 64) {
    copy16B_g2l(&qs[i * 8], &qb[(size_t)n * (HH * EE) + i * 8]);
    copy16B_g2l(&ks[i * 8], &kb[(size_t)n * (HH * EE) + i * 8]);
  }
  copy_fence();
  __syncthreads();
  int h = tid >> 3, g = tid & 7;
  float acc = 0.f;
  #pragma unroll
  for (int c = 0; c < 16; c++) {
    v8h_t qv = *(const v8h_t*)&qs[h * EE + c * 8];
    v8h_t kv = *(const v8h_t*)&ks[g * EE + c * 8];
    #pragma unroll
    for (int j = 0; j < 8; j++) acc += (float)qv[j] * (float)kv[j];
  }
  scores[(size_t)n * 64 + tid] = acc * 0.08838834764831843f;  // 1/sqrt(128)
}

// ---------------------------------------------------------------------------
// K3: segment boundaries (batch is sorted) + per-segment softmax stats
// ---------------------------------------------------------------------------
__global__ void seg_init(int* segStart, int* segEnd) {
  int g = blockIdx.x * blockDim.x + threadIdx.x;
  if (g < GG) { segStart[g] = 0; segEnd[g] = 0; }
}

__global__ void seg_bounds(const int* __restrict__ batch, int* segStart,
                           int* segEnd, int n_total) {
  int n = blockIdx.x * blockDim.x + threadIdx.x;
  if (n >= n_total) return;
  int b = batch[n];
  if (n == 0 || batch[n - 1] != b) segStart[b] = n;
  if (n == n_total - 1 || batch[n + 1] != b) segEnd[b] = n + 1;
}

__global__ __launch_bounds__(256)
void seg_softmax_stats(const float* __restrict__ scores, const int* __restrict__ segStart,
                       const int* __restrict__ segEnd, float* __restrict__ m,
                       float* __restrict__ den) {
  int g = blockIdx.x;
  int s = segStart[g], e = segEnd[g];
  int chan = threadIdx.x & 63;
  int sub  = threadIdx.x >> 6;
  __shared__ float red[256];
  __shared__ float mchan[64];
  float mx = -3.4e38f;
  for (int n = s + sub; n < e; n += 4)
    mx = fmaxf(mx, scores[(size_t)n * 64 + chan]);
  red[threadIdx.x] = mx;
  __syncthreads();
  if (sub == 0) {
    mx = fmaxf(fmaxf(red[chan], red[chan + 64]),
               fmaxf(red[chan + 128], red[chan + 192]));
    mchan[chan] = mx;
    m[g * 64 + chan] = mx;
  }
  __syncthreads();
  float mm = mchan[chan];
  float sum = 0.f;
  for (int n = s + sub; n < e; n += 4)
    sum += __expf(scores[(size_t)n * 64 + chan] - mm);
  red[threadIdx.x] = sum;
  __syncthreads();
  if (sub == 0)
    den[g * 64 + chan] = red[chan] + red[chan + 64] + red[chan + 128] + red[chan + 192];
}

// ---------------------------------------------------------------------------
// K4: aw = softmax(scores), s = aw @ v  (vectorized VALU, v async-staged in
// LDS halves so the copy overlaps aw computation), then out-MLP via WMMA:
//     y = relu(s[16,1024] @ Wo1) ; out = y @ Wo2 + bo2 + para  (f32)
// grid = N/16, 256 threads = 8 waves.  Static LDS = 60 KB.
// ---------------------------------------------------------------------------
__device__ __forceinline__ void awv_half(const float* aw, const _Float16* vtile,
                                         _Float16* sh, int tid, int nlbase) {
  // tasks: 8 nodes x 8 h x 16 e-groups(8-wide) = 1024; 4 per thread
  for (int t = tid; t < 1024; t += 256) {
    int nl  = t >> 7;             // 0..7 (local)
    int rem = t & 127;
    int h   = rem >> 4;
    int e0  = (rem & 15) << 3;
    const float* awrow = &aw[(nlbase + nl) * 64 + h * 8];
    float accv[8];
    #pragma unroll
    for (int j = 0; j < 8; j++) accv[j] = 0.f;
    #pragma unroll
    for (int g = 0; g < 8; g++) {
      v8h_t vv = *(const v8h_t*)&vtile[nl * 1024 + g * EE + e0];
      float a = awrow[g];
      #pragma unroll
      for (int j = 0; j < 8; j++) accv[j] += a * (float)vv[j];
    }
    v8h_t o;
    #pragma unroll
    for (int j = 0; j < 8; j++) o[j] = (_Float16)accv[j];
    *(v8h_t*)&sh[(nlbase + nl) * 1024 + h * EE + e0] = o;
  }
}

__global__ __launch_bounds__(256)
void attn_out_kernel(const _Float16* __restrict__ vb, const float* __restrict__ scores,
                     const int* __restrict__ batch, const float* __restrict__ m,
                     const float* __restrict__ den,
                     const _Float16* __restrict__ wo1t, const float* __restrict__ bo1,
                     const _Float16* __restrict__ wo2t, const float* __restrict__ bo2,
                     const float* __restrict__ para, float* __restrict__ out) {
  __shared__ float    aw[16 * 64];      // 4 KB
  __shared__ _Float16 sh[16 * 1024];    // 32 KB
  __shared__ _Float16 yh[16 * 256];     // 8 KB
  __shared__ _Float16 vtile[8 * 1024];  // 16 KB (half of the 16-node v tile)
  int tid = threadIdx.x;
  int n0  = blockIdx.x * 16;

  // kick off async copy of v for nodes 0..7 (1024 x 16B chunks)
  for (int i = tid; i < 8 * 128; i += 256) {
    int nl = i >> 7, c8 = (i & 127) << 3;
    copy16B_g2l(&vtile[nl * 1024 + c8], &vb[((size_t)(n0 + nl)) * (HH * EE) + c8]);
  }

  // softmax weights (overlaps the async copy above)
  for (int i = tid; i < 16 * 64; i += 256) {
    int nl = i >> 6, c = i & 63;
    int n = n0 + nl;
    int b = batch[n];
    aw[i] = __expf(scores[(size_t)n * 64 + c] - m[b * 64 + c]) /
            (den[b * 64 + c] + 1e-16f);
  }
  copy_fence();
  __syncthreads();

  // s = aw @ v for nodes 0..7
  awv_half(aw, vtile, sh, tid, 0);
  __syncthreads();  // all reads of vtile done

  // second half: nodes 8..15
  for (int i = tid; i < 8 * 128; i += 256) {
    int nl = i >> 7, c8 = (i & 127) << 3;
    copy16B_g2l(&vtile[nl * 1024 + c8], &vb[((size_t)(n0 + 8 + nl)) * (HH * EE) + c8]);
  }
  copy_fence();
  __syncthreads();

  awv_half(aw, vtile, sh, tid, 8);
  __syncthreads();

  int w = tid >> 5, lane = tid & 31, l15 = lane & 15;

  // O1: [16,1024] @ Wo1t(256x1024 [out,in]) -> yh[16,256], relu
  for (int j = 0; j < 2; j++) {
    int col0 = (w * 2 + j) * 16;
    float bias = bo1[col0 + l15];
    v8f acc;
    #pragma unroll
    for (int i = 0; i < 8; i++) acc[i] = bias;
    #pragma unroll 4
    for (int kk = 0; kk < 32; kk++) {
      v16h a = load_frag(&sh[kk * 32], 1024);
      v16h b = load_frag(&wo1t[(size_t)col0 * 1024 + kk * 32], 1024);
      acc = wmma_f16(a, b, acc);
    }
    store_tile_f16(&yh[col0], 256, acc, true);
  }
  __syncthreads();

  // O2: [16,256] @ Wo2t(128x256) -> f32 out (+bo2 +para)
  {
    int col0 = w * 16;
    float bias = bo2[col0 + l15] + para[col0 + l15];
    v8f acc;
    #pragma unroll
    for (int i = 0; i < 8; i++) acc[i] = bias;
    #pragma unroll
    for (int kk = 0; kk < 8; kk++) {
      v16h a = load_frag(&yh[kk * 32], 256);
      v16h b = load_frag(&wo2t[col0 * 256 + kk * 32], 256);
      acc = wmma_f16(a, b, acc);
    }
    int col = lane & 15, rbase = (lane >> 4) * 8;
    #pragma unroll
    for (int r = 0; r < 8; r++)
      out[((size_t)(n0 + rbase + r)) * EE + col0 + col] = acc[r];
  }
}

// ---------------------------------------------------------------------------
extern "C" void kernel_launch(void* const* d_in, const int* in_sizes, int n_in,
                              void* d_out, int out_size, void* d_ws, size_t ws_size,
                              hipStream_t stream) {
  const float* score = (const float*)d_in[0];
  const int*   batch = (const int*)d_in[2];
  const float* pos   = (const float*)d_in[3];
  const float* para  = (const float*)d_in[4];
  const float* Wq1 = (const float*)d_in[5];  const float* bq1 = (const float*)d_in[6];
  const float* Wq2 = (const float*)d_in[7];  const float* bq2 = (const float*)d_in[8];
  const float* Wk1 = (const float*)d_in[9];  const float* bk1 = (const float*)d_in[10];
  const float* Wk2 = (const float*)d_in[11]; const float* bk2 = (const float*)d_in[12];
  const float* Wv1 = (const float*)d_in[13]; const float* bv1 = (const float*)d_in[14];
  const float* Wv2 = (const float*)d_in[15]; const float* bv2 = (const float*)d_in[16];
  const float* Wo1 = (const float*)d_in[17]; const float* bo1 = (const float*)d_in[18];
  const float* Wo2 = (const float*)d_in[19]; const float* bo2 = (const float*)d_in[20];
  float* out = (float*)d_out;

  char* ws = (char*)d_ws;
  size_t off = 0;
  auto alloc = [&](size_t bytes) -> void* {
    void* p = ws + off;
    off += (bytes + 255) & ~(size_t)255;
    return p;
  };

  _Float16* smh  = (_Float16*)alloc((size_t)NN * DD * 2);
  _Float16* w1q  = (_Float16*)alloc((size_t)HH * HIDW * DD * 2);
  _Float16* w1k  = (_Float16*)alloc((size_t)HH * HIDW * DD * 2);
  _Float16* w1v  = (_Float16*)alloc((size_t)HH * HIDW * DD * 2);
  _Float16* w2q  = (_Float16*)alloc((size_t)HH * EE * HIDW * 2);
  _Float16* w2k  = (_Float16*)alloc((size_t)HH * EE * HIDW * 2);
  _Float16* w2v  = (_Float16*)alloc((size_t)HH * EE * HIDW * 2);
  _Float16* wo1t = (_Float16*)alloc((size_t)HIDW * (HH * EE) * 2);
  _Float16* wo2t = (_Float16*)alloc((size_t)EE * HIDW * 2);
  _Float16* qb   = (_Float16*)alloc((size_t)NN * HH * EE * 2);
  _Float16* kb   = (_Float16*)alloc((size_t)NN * HH * EE * 2);
  _Float16* vb   = (_Float16*)alloc((size_t)NN * HH * EE * 2);
  float*    scb  = (float*)alloc((size_t)NN * 64 * 4);
  float*    mbuf = (float*)alloc((size_t)GG * 64 * 4);
  float*    dbuf = (float*)alloc((size_t)GG * 64 * 4);
  int*      segS = (int*)alloc((size_t)GG * 4);
  int*      segE = (int*)alloc((size_t)GG * 4);

  // K0: activations
  prep_sm<<<(NN * DD / 8 + 255) / 256, 256, 0, stream>>>(score, pos, smh, NN * DD / 8);

  // K0b: weights -> f16 transposed [out,in]
  auto tr = [&](const float* in, _Float16* o, int B, int R, int C) {
    int total = B * R * C;
    cvt_transpose<<<(total + 255) / 256, 256, 0, stream>>>(in, o, B, R, C);
  };
  tr(Wq1, w1q, HH, DD, HIDW);   tr(Wk1, w1k, HH, DD, HIDW);   tr(Wv1, w1v, HH, DD, HIDW);
  tr(Wq2, w2q, HH, HIDW, EE);   tr(Wk2, w2k, HH, HIDW, EE);   tr(Wv2, w2v, HH, HIDW, EE);
  tr(Wo1, wo1t, 1, HH * EE, HIDW);
  tr(Wo2, wo2t, 1, HIDW, EE);

  // K1: q,k,v projections (the 1.5e11-MAC GEMMs)
  qkv_kernel<<<dim3(NN / 64, 24), 256, 0, stream>>>(
      smh, w1q, w1k, w1v, bq1, bk1, bv1, w2q, w2k, w2v, bq2, bk2, bv2, qb, kb, vb);

  // K2: per-node attention scores
  scores_kernel<<<NN, 64, 0, stream>>>(qb, kb, scb);

  // K3: segment softmax stats
  seg_init<<<1, GG, 0, stream>>>(segS, segE);
  seg_bounds<<<(NN + 255) / 256, 256, 0, stream>>>(batch, segS, segE, NN);
  seg_softmax_stats<<<GG, 256, 0, stream>>>(scb, segS, segE, mbuf, dbuf);

  // K4: attention apply + out MLP
  attn_out_kernel<<<NN / 16, 256, 0, stream>>>(
      vb, scb, batch, mbuf, dbuf, wo1t, bo1, wo2t, bo2, para, out);

  (void)in_sizes; (void)n_in; (void)out_size; (void)ws_size;
}
